// TorchNeighbors_68247030334237
// MI455X (gfx1250) — compile-verified, matrix operand order checked by reading, not run on hardware
//
#include <hip/hip_runtime.h>
#include <math.h>

typedef __attribute__((ext_vector_type(16))) _Float16 v16h;
typedef __attribute__((ext_vector_type(8)))  float    v8f;

#define NMOL   16
#define NATOM  384
#define NOFFS  27
#define NTILE  24                       // 384/16 j- and i-tiles
#define ROWS   (NMOL*NOFFS*NATOM)       // 165888 output rows (b,o,i)
#define UNITS  (NMOL*NOFFS*NTILE)       // 10368 wave-units (b,o,itile)
#define MAXP   400000
#define CUTF   5.5f
#define PRETH  33.0f   // 30.25 + conservative bound on f16 Gram-matrix error (~1.8)

static_assert(ROWS == 162*1024, "scan sizing assumption");

// ---------------- exact mask (must match reference fp behavior) -------------
__device__ __forceinline__ bool exact_mask(float4 pi, float4 pj,
                                           float pmx, float pmy, float pmz,
                                           int o, int i, int j) {
#pragma clang fp contract(off)
  float dx = (pi.x - pj.x) + pmx;
  float dy = (pi.y - pj.y) + pmy;
  float dz = (pi.z - pj.z) + pmz;
  float s  = (dx*dx + dy*dy) + dz*dz;
  return (sqrtf(s) < CUTF) && (o != 13 || i != j);
}

// ---------------- kernel 0: zero outputs + misc -----------------------------
__global__ void tn_zero(float* out, int* misc) {
  long long idx    = (long long)blockIdx.x * blockDim.x + threadIdx.x;
  long long stride = (long long)gridDim.x * blockDim.x;
  const long long total = 10LL * MAXP;
  for (long long t = idx; t < total; t += stride) out[t] = 0.0f;
  if (idx == 0) { misc[0] = 0; misc[1] = 0; }
}

// ---------------- kernel 1: wrapped coords + wrap offsets -------------------
__global__ __launch_bounds__(256) void tn_prep(const float* __restrict__ coords,
                                               const float* __restrict__ cell,
                                               float4* __restrict__ wc,
                                               int4*  __restrict__ wo) {
#pragma clang fp contract(off)
  int t = blockIdx.x * blockDim.x + threadIdx.x;
  if (t >= NMOL * NATOM) return;
  int b = t / NATOM;
  const float* cb = cell + b * 9;
  float a00=cb[0],a01=cb[1],a02=cb[2],a10=cb[3],a11=cb[4],a12=cb[5],a20=cb[6],a21=cb[7],a22=cb[8];
  // adjugate inverse (exact for the diagonal harness cell; matches jnp.linalg.inv there)
  float c00 = a11*a22 - a12*a21;
  float c01 = a12*a20 - a10*a22;
  float c02 = a10*a21 - a11*a20;
  float det = a00*c00 + a01*c01 + a02*c02;
  float i00=c00/det, i10=c01/det, i20=c02/det;
  float i01=(a02*a21 - a01*a22)/det, i11=(a00*a22 - a02*a20)/det, i21=(a01*a20 - a00*a21)/det;
  float i02=(a01*a12 - a02*a11)/det, i12=(a02*a10 - a00*a12)/det, i22=(a00*a11 - a01*a10)/det;
  const float* cd = coords + (long long)t * 3;
  float x = cd[0], y = cd[1], z = cd[2];
  float pe0 = x*i00 + y*i10 + z*i20;   // proj[e] = sum_d coord[d]*inv[d][e]
  float pe1 = x*i01 + y*i11 + z*i21;
  float pe2 = x*i02 + y*i12 + z*i22;
  float fl0 = floorf(pe0), fl1 = floorf(pe1), fl2 = floorf(pe2);
  float f0 = pe0 - fl0, f1 = pe1 - fl1, f2 = pe2 - fl2;
  float wx = f0*a00 + f1*a10 + f2*a20; // wc[d] = sum_e frac[e]*cell[e][d]
  float wy = f0*a01 + f1*a11 + f2*a21;
  float wz = f0*a02 + f1*a12 + f2*a22;
  wc[t] = make_float4(wx, wy, wz, 0.0f);
  wo[t] = make_int4((int)fl0, (int)fl1, (int)fl2, 0);
}

// -------- WMMA A-fragment for one i-tile: a = [-2p, |p|^2, 1, 0...] ---------
__device__ __forceinline__ v16h build_A(const float4* wcb, int it, int lane) {
  v16h A = {};
  float4 p = wcb[it * 16 + (lane & 15)];
  _Float16 hx = (_Float16)p.x, hy = (_Float16)p.y, hz = (_Float16)p.z;
  float fx = (float)hx, fy = (float)hy, fz = (float)hz;
  float sp = fx*fx + fy*fy + fz*fz;
  if (lane < 16) {
    A[0] = (_Float16)(-2.0f * fx);
    A[1] = (_Float16)(-2.0f * fy);
    A[2] = (_Float16)(-2.0f * fz);
    A[3] = (_Float16)sp;
    A[4] = (_Float16)1.0f;
  }
  return A;
}

// -------- WMMA B-fragment for one j-tile: b = [q, 1, |q|^2, 0...] -----------
__device__ __forceinline__ v16h build_B(float4 q4, float pmx, float pmy, float pmz, int lane) {
  v16h Bf = {};
  float qx = q4.x - pmx, qy = q4.y - pmy, qz = q4.z - pmz;  // q = wc_j - perm
  _Float16 hqx = (_Float16)qx, hqy = (_Float16)qy, hqz = (_Float16)qz;
  float gx = (float)hqx, gy = (float)hqy, gz = (float)hqz;
  float sq = gx*gx + gy*gy + gz*gz;
  if (lane < 16) {
    Bf[0] = hqx; Bf[1] = hqy; Bf[2] = hqz;
    Bf[3] = (_Float16)1.0f;
    Bf[4] = (_Float16)sq;
  }
  return Bf;
}

// ---------------- kernel 2: per-row pair counts (WMMA prefilter) ------------
__global__ __launch_bounds__(32) void tn_count(const float4* __restrict__ wc,
                                               const float* __restrict__ cell,
                                               int* __restrict__ rowCnt) {
#pragma clang fp contract(off)
  int u = blockIdx.x, lane = threadIdx.x;
  int b  = u / (NOFFS * NTILE);
  int r  = u % (NOFFS * NTILE);
  int o  = r / NTILE;
  int it = r % NTILE;
  float ox = (float)(o / 9 - 1), oy = (float)((o / 3) % 3 - 1), oz = (float)(o % 3 - 1);
  const float* cb = cell + b * 9;
  float pmx = ox*cb[0] + oy*cb[3] + oz*cb[6];
  float pmy = ox*cb[1] + oy*cb[4] + oz*cb[7];
  float pmz = ox*cb[2] + oy*cb[5] + oz*cb[8];
  const float4* wcb = wc + b * NATOM;

  v16h A = build_A(wcb, it, lane);
  int cnt[16];
#pragma unroll
  for (int m = 0; m < 16; ++m) cnt[m] = 0;

  for (int jt = 0; jt < NTILE; ++jt) {
    if (jt + 1 < NTILE) __builtin_prefetch(wcb + (jt + 1) * 16, 0, 1);
    float4 q4 = wcb[jt * 16 + (lane & 15)];
    v16h Bf = build_B(q4, pmx, pmy, pmz, lane);
    v8f c = {};
    c = __builtin_amdgcn_wmma_f32_16x16x32_f16(false, A, false, Bf,
                                               (short)0, c, false, false);
#pragma unroll
    for (int v = 0; v < 8; ++v) {
      bool pre = c[v] < PRETH;
      unsigned bm = __builtin_amdgcn_ballot_w32(pre);
      unsigned m = 0u;
      if (bm) {
        bool ex = false;
        if (pre) {
          int M = (lane < 16) ? v : v + 8;
          int i = it * 16 + M;
          int j = jt * 16 + (lane & 15);
          float4 pi = wcb[i];
          ex = exact_mask(pi, q4, pmx, pmy, pmz, o, i, j);
        }
        m = __builtin_amdgcn_ballot_w32(ex);
      }
      cnt[v]     += __popc(m & 0xFFFFu);
      cnt[v + 8] += __popc(m >> 16);
    }
  }
  if (lane == 0) {
    int row = (b * NOFFS + o) * NATOM + it * 16;
#pragma unroll
    for (int m = 0; m < 16; ++m) rowCnt[row + m] = cnt[m];
  }
}

// ---------------- kernel 3a: in-place block scan (1024 rows / block) --------
__global__ __launch_bounds__(256) void tn_scan1(int* __restrict__ rowCnt,
                                                int* __restrict__ chunkSum) {
  __shared__ int sh[256];
  int t = threadIdx.x;
  int base = blockIdx.x * 1024 + t * 4;
  int v0 = rowCnt[base], v1 = rowCnt[base+1], v2 = rowCnt[base+2], v3 = rowCnt[base+3];
  int s = v0 + v1 + v2 + v3;
  sh[t] = s; __syncthreads();
  for (int d = 1; d < 256; d <<= 1) {
    int x = (t >= d) ? sh[t - d] : 0;
    __syncthreads();
    sh[t] += x;
    __syncthreads();
  }
  int excl = sh[t] - s;
  rowCnt[base]     = excl;
  rowCnt[base + 1] = excl + v0;
  rowCnt[base + 2] = excl + v0 + v1;
  rowCnt[base + 3] = excl + v0 + v1 + v2;
  if (t == 255) chunkSum[blockIdx.x] = sh[t];
}

// ---------------- kernel 3b: scan the 162 chunk sums ------------------------
__global__ __launch_bounds__(256) void tn_scan2(int* __restrict__ chunkSum) {
  __shared__ int sh[256];
  int t = threadIdx.x;
  int v = (t < 162) ? chunkSum[t] : 0;
  sh[t] = v; __syncthreads();
  for (int d = 1; d < 256; d <<= 1) {
    int x = (t >= d) ? sh[t - d] : 0;
    __syncthreads();
    sh[t] += x;
    __syncthreads();
  }
  if (t < 162) chunkSum[t] = sh[t] - v;  // exclusive
}

// ---------------- kernel 4: ordered fill (WMMA prefilter again) -------------
__global__ __launch_bounds__(32) void tn_fill(const float4* __restrict__ wc,
                                              const int4*  __restrict__ wo,
                                              const float* __restrict__ coords,
                                              const float* __restrict__ cell,
                                              const int*   __restrict__ invReal,
                                              const int*   __restrict__ rowOff,
                                              const int*   __restrict__ chunkOff,
                                              float* __restrict__ out,
                                              int*   __restrict__ misc) {
#pragma clang fp contract(off)
  int u = blockIdx.x, lane = threadIdx.x;
  int b  = u / (NOFFS * NTILE);
  int r  = u % (NOFFS * NTILE);
  int o  = r / NTILE;
  int it = r % NTILE;
  int iox = o / 9 - 1, ioy = (o / 3) % 3 - 1, ioz = o % 3 - 1;
  float ox = (float)iox, oy = (float)ioy, oz = (float)ioz;
  const float* cb = cell + b * 9;
  float pmx = ox*cb[0] + oy*cb[3] + oz*cb[6];
  float pmy = ox*cb[1] + oy*cb[4] + oz*cb[7];
  float pmz = ox*cb[2] + oy*cb[5] + oz*cb[8];
  const float4* wcb = wc + b * NATOM;
  const int4*  wob  = wo + b * NATOM;

  int row = (b * NOFFS + o) * NATOM + it * 16;
  int cbase = chunkOff[row >> 10];
  int pos[16];
#pragma unroll
  for (int m = 0; m < 16; ++m) pos[m] = rowOff[row + m] + cbase;

  v16h A = build_A(wcb, it, lane);

  for (int jt = 0; jt < NTILE; ++jt) {
    if (jt + 1 < NTILE) __builtin_prefetch(wcb + (jt + 1) * 16, 0, 1);
    float4 q4 = wcb[jt * 16 + (lane & 15)];
    v16h Bf = build_B(q4, pmx, pmy, pmz, lane);
    v8f c = {};
    c = __builtin_amdgcn_wmma_f32_16x16x32_f16(false, A, false, Bf,
                                               (short)0, c, false, false);
#pragma unroll
    for (int v = 0; v < 8; ++v) {
      bool pre = c[v] < PRETH;
      unsigned bm = __builtin_amdgcn_ballot_w32(pre);
      int M = (lane < 16) ? v : v + 8;
      int i = it * 16 + M;
      int j = jt * 16 + (lane & 15);
      unsigned m = 0u;
      bool ex = false;
      if (bm) {
        if (pre) {
          float4 pi = wcb[i];
          ex = exact_mask(pi, q4, pmx, pmy, pmz, o, i, j);
        }
        m = __builtin_amdgcn_ballot_w32(ex);
      }
      unsigned lo = m & 0xFFFFu, hi = m >> 16;
      if (ex) {
        int rank = (lane < 16) ? __popc(lo & ((1u << lane) - 1u))
                               : __popc(hi & ((1u << (lane - 16)) - 1u));
        int p = ((lane < 16) ? pos[v] : pos[v + 8]) + rank;
        if (p < MAXP) {
          int4 wi = wob[i], wj = wob[j];
          int offX = iox - (wi.x - wj.x);
          int offY = ioy - (wi.y - wj.y);
          int offZ = ioz - (wi.z - wj.z);
          int ma = abs(offX); ma = max(ma, abs(offY)); ma = max(ma, abs(offZ));
          if (ma > 0) atomicMax(&misc[0], ma);
          float fx = (float)offX, fy = (float)offY, fz = (float)offZ;
          float px = fx*cb[0] + fy*cb[3] + fz*cb[6];  // offsets @ cell
          float py = fx*cb[1] + fy*cb[4] + fz*cb[7];
          float pz = fx*cb[2] + fy*cb[5] + fz*cb[8];
          const float* ci = coords + (long long)(b * NATOM + i) * 3;
          const float* cj = coords + (long long)(b * NATOM + j) * 3;
          float pcx = ci[0] - cj[0] + px;
          float pcy = ci[1] - cj[1] + py;
          float pcz = ci[2] - cj[2] + pz;
          float s = (pcx*pcx + pcy*pcy) + pcz*pcz;
          out[p]            = sqrtf(s);
          out[MAXP + p]     = (float)invReal[b * NATOM + i];
          out[2 * MAXP + p] = (float)invReal[b * NATOM + j];
          out[3 * MAXP + 3 * p + 0] = pcx;
          out[3 * MAXP + 3 * p + 1] = pcy;
          out[3 * MAXP + 3 * p + 2] = pcz;
          out[6 * MAXP + 3 * p + 0] = fx;
          out[6 * MAXP + 3 * p + 1] = fy;
          out[6 * MAXP + 3 * p + 2] = fz;
        }
      }
      pos[v]     += __popc(lo);
      pos[v + 8] += __popc(hi);
    }
  }
}

// ---------------- kernel 5: offset_index (needs global max_images) ----------
__global__ void tn_offidx(const int* __restrict__ misc, float* __restrict__ out) {
  int p = blockIdx.x * blockDim.x + threadIdx.x;
  if (p >= MAXP) return;
  int mi = misc[0];
  int noff = 2 * mi + 1;
  int o0 = (int)out[6 * MAXP + 3 * p + 0] + mi;
  int o1 = (int)out[6 * MAXP + 3 * p + 1] + mi;
  int o2 = (int)out[6 * MAXP + 3 * p + 2] + mi;
  out[9 * MAXP + p] = (float)(o2 + noff * (o1 + noff * o0));
}

// ---------------- host entry ------------------------------------------------
extern "C" void kernel_launch(void* const* d_in, const int* in_sizes, int n_in,
                              void* d_out, int out_size, void* d_ws, size_t ws_size,
                              hipStream_t stream) {
  (void)in_sizes; (void)n_in; (void)out_size; (void)ws_size;
  const float* coords  = (const float*)d_in[0];
  const int*   invReal = (const int*)d_in[3];
  const float* cell    = (const float*)d_in[4];
  float* out = (float*)d_out;

  char* ws = (char*)d_ws;
  const size_t WS_WC   = 0;
  const size_t WS_WO   = WS_WC  + sizeof(float4) * NMOL * NATOM;   //  98304
  const size_t WS_CNT  = WS_WO  + sizeof(int4)   * NMOL * NATOM;   // 196608
  const size_t WS_CHK  = WS_CNT + sizeof(int) * ROWS;              // 860160
  const size_t WS_MISC = WS_CHK + sizeof(int) * 256;               // 861184
  float4* wc     = (float4*)(ws + WS_WC);
  int4*   wo     = (int4*)  (ws + WS_WO);
  int*    rowCnt = (int*)   (ws + WS_CNT);
  int*    chunk  = (int*)   (ws + WS_CHK);
  int*    misc   = (int*)   (ws + WS_MISC);

  tn_zero <<<2048, 256, 0, stream>>>(out, misc);
  tn_prep <<<(NMOL * NATOM) / 256, 256, 0, stream>>>(coords, cell, wc, wo);
  tn_count<<<UNITS, 32, 0, stream>>>(wc, cell, rowCnt);
  tn_scan1<<<ROWS / 1024, 256, 0, stream>>>(rowCnt, chunk);
  tn_scan2<<<1, 256, 0, stream>>>(chunk);
  tn_fill <<<UNITS, 32, 0, stream>>>(wc, wo, coords, cell, invReal,
                                     rowCnt, chunk, out, misc);
  tn_offidx<<<(MAXP + 255) / 256, 256, 0, stream>>>(misc, out);
}
